// SingleScaleDynamics_39797166964927
// MI455X (gfx1250) — compile-verified, matrix operand order checked by reading, not run on hardware
//
#include <hip/hip_runtime.h>
#include <hip/hip_bf16.h>

typedef __bf16 bf16;
typedef __attribute__((ext_vector_type(16))) __bf16 v16bf;
typedef __attribute__((ext_vector_type(8)))  __bf16 v8bf;
typedef __attribute__((ext_vector_type(8)))  float  v8f;

#define AS1 __attribute__((address_space(1)))

#define HDIM 512
#define BATCH 4096
#define TSTEPS 64
#define ROWS 32          // batch rows per workgroup
#define NTHREADS 256     // 8 wave32
#define KTILES 16        // 512 / 32
#define NTILES_W 96      // 1536 / 16

// ---------------------------------------------------------------------------
// Pack w_ih / w_hh (f32 row-major [1536,512]) into bf16 tiles laid out exactly
// as the WMMA B-matrix VGPR fragment (16x16x32 bf16):
//   wpack[mat][ntile][ktile][lane][e], e in 0..15
//   column n  = ntile*16 + (lane & 15)       (row of W)
//   K within tile = (e<8 ? e : 16+(e-8)) + (lane>=16 ? 8 : 0)
// Each lane of the GEMM then loads its whole fragment as 32 contiguous bytes.
// ---------------------------------------------------------------------------
__global__ void gru_pack_weights(const float* __restrict__ w_ih,
                                 const float* __restrict__ w_hh,
                                 bf16* __restrict__ wpack) {
  const size_t total = (size_t)2 * NTILES_W * KTILES * 32 * 16;
  size_t idx = (size_t)blockIdx.x * blockDim.x + threadIdx.x;
  if (idx >= total) return;
  int e     = (int)(idx & 15);
  int lane  = (int)((idx >> 4) & 31);
  int kt    = (int)((idx >> 9) & 15);
  int ntile = (int)((idx >> 13) % NTILES_W);
  int mat   = (int)((idx >> 13) / NTILES_W);
  int n  = ntile * 16 + (lane & 15);
  int kk = (e < 8 ? e : 16 + (e - 8)) + ((lane >= 16) ? 8 : 0);
  int K  = kt * 32 + kk;
  const float* w = mat ? w_hh : w_ih;
  wpack[idx] = (bf16)w[(size_t)n * HDIM + K];
}

__device__ __forceinline__ float sigmoid_f(float x) {
  return 1.0f / (1.0f + __expf(-x));
}
__device__ __forceinline__ float tanh_f(float x) {
  float t = __expf(-2.0f * x);
  return (1.0f - t) / (1.0f + t);
}

// ---------------------------------------------------------------------------
// One workgroup owns ROWS=32 batch rows and iterates the full 64-step
// recurrence with h resident in LDS (bf16). Per step, per wave:
//   wave w: mi = w&1 selects 16-row tile, ng = w>>1 selects 128-col group.
//   4 tasks of 32 columns; each task accumulates 12 C-tiles
//   (3 gates x {ih,hh} x 2 n-tiles) over K=512 via v_wmma_f32_16x16x32_bf16.
// Weight fragments are re-streamed from L2 every step. The base pointer is
// laundered through an empty asm as an address_space(1) pointer: opaque to
// LICM (no hoist/spill across the 64-step loop) but still provably global,
// so loads lower to global_load_b128 (saddr + imm offset), not flat_load.
// ---------------------------------------------------------------------------
__global__ __launch_bounds__(NTHREADS)
void gru_recurrence_kernel(const float* __restrict__ z_init,
                           const float* __restrict__ b_ih,
                           const float* __restrict__ b_hh,
                           const bf16* __restrict__ wpack,
                           float* __restrict__ out) {
  __shared__ bf16  sh[ROWS][HDIM];       // 32 KB: current h (bf16)
  __shared__ float sbias[2][3 * HDIM];   // 12 KB: b_ih, b_hh

  const int tid    = threadIdx.x;
  const int lane   = tid & 31;
  const int wave   = tid >> 5;
  const int rbase  = blockIdx.x * ROWS;
  const int laneHi = lane >> 4;          // 0 or 1
  const int lmod   = lane & 15;
  const int mi     = wave & 1;           // row-tile (16 rows)
  const int ng     = wave >> 1;          // 128-column group

  for (int i = tid; i < 3 * HDIM; i += NTHREADS) {
    sbias[0][i] = b_ih[i];
    sbias[1][i] = b_hh[i];
  }
  for (int i = tid; i < ROWS * HDIM; i += NTHREADS) {
    int r = i >> 9, c = i & (HDIM - 1);
    sh[r][c] = (bf16)z_init[(size_t)(rbase + r) * HDIM + c];
  }
  __syncthreads();

  for (int t = 0; t < TSTEPS; ++t) {
    float hnew[4][2][8];  // [task][ntile][row-within-half]

    // ---- phase 1: GEMMs + gate math (reads sh only) ----
    #pragma unroll
    for (int c = 0; c < 4; ++c) {
      const int cbase = ng * 128 + c * 32;

      // Opaque global-AS pointer: not hoistable across t, still global_load.
      const AS1 bf16* wb = (const AS1 bf16*)wpack;
      asm volatile("" : "+s"(wb));

      v8f acc[12];
      #pragma unroll
      for (int i = 0; i < 12; ++i) acc[i] = {};

      for (int kt = 0; kt < KTILES; ++kt) {
        // A fragment from LDS: lane holds row m, K = {0..7,16..23}(+8 hi half)
        const int m  = mi * 16 + lmod;
        const int kb = kt * 32 + laneHi * 8;
        v8bf a_lo = *(const v8bf*)&sh[m][kb];
        v8bf a_hi = *(const v8bf*)&sh[m][kb + 16];
        v16bf A = __builtin_shufflevector(a_lo, a_hi,
            0, 1, 2, 3, 4, 5, 6, 7, 8, 9, 10, 11, 12, 13, 14, 15);

        #pragma unroll
        for (int gm = 0; gm < 6; ++gm) {      // (gate r,z,n) x (ih,hh)
          const int g   = gm >> 1;
          const int mat = gm & 1;
          #pragma unroll
          for (int nt = 0; nt < 2; ++nt) {
            const int ntile = g * 32 + (cbase >> 4) + nt;
            const AS1 bf16* bp = wb +
                ((((size_t)mat * NTILES_W + ntile) * KTILES + kt) * 32 + lane) * 16;
            v16bf Bt = *(const AS1 v16bf*)bp;
            acc[gm * 2 + nt] = __builtin_amdgcn_wmma_f32_16x16x32_bf16(
                false, A, false, Bt, (short)0, acc[gm * 2 + nt], false, false);
          }
        }
      }

      // gate math in f32; C layout: elem i -> row (i + 8*laneHi), col lmod
      #pragma unroll
      for (int nt = 0; nt < 2; ++nt) {
        const int j   = cbase + nt * 16 + lmod;
        const float br  = sbias[0][j] + sbias[1][j];
        const float bz  = sbias[0][HDIM + j] + sbias[1][HDIM + j];
        const float bin = sbias[0][2 * HDIM + j];
        const float bhn = sbias[1][2 * HDIM + j];
        #pragma unroll
        for (int i = 0; i < 8; ++i) {
          const int m = mi * 16 + i + 8 * laneHi;
          const float r  = sigmoid_f(acc[0 + nt][i] + acc[2 + nt][i] + br);
          const float zz = sigmoid_f(acc[4 + nt][i] + acc[6 + nt][i] + bz);
          const float nn = tanh_f(acc[8 + nt][i] + bin +
                                  r * (acc[10 + nt][i] + bhn));
          const float hold = (float)sh[m][j];
          const float hv   = (1.0f - zz) * nn + zz * hold;
          hnew[c][nt][i] = hv;
          // Streaming output (written once, never re-read): non-temporal so
          // the 537 MB result stream does not evict L2-resident weights.
          __builtin_nontemporal_store(
              hv, &out[((size_t)(rbase + m) * TSTEPS + t) * HDIM + j]);
        }
      }
    }

    __syncthreads();  // all reads of sh for step t complete

    // ---- phase 2: write h_{t+1} back to LDS ----
    #pragma unroll
    for (int c = 0; c < 4; ++c) {
      const int cbase = ng * 128 + c * 32;
      #pragma unroll
      for (int nt = 0; nt < 2; ++nt) {
        const int j = cbase + nt * 16 + lmod;
        #pragma unroll
        for (int i = 0; i < 8; ++i) {
          const int m = mi * 16 + i + 8 * laneHi;
          sh[m][j] = (bf16)hnew[c][nt][i];
        }
      }
    }

    __syncthreads();  // h_{t+1} visible to all waves
  }

  if (blockIdx.x == 0 && tid == 0) {
    out[(size_t)BATCH * TSTEPS * HDIM] = 0.0f;  // trailing scalar output
  }
}

extern "C" void kernel_launch(void* const* d_in, const int* in_sizes, int n_in,
                              void* d_out, int out_size, void* d_ws, size_t ws_size,
                              hipStream_t stream) {
  (void)in_sizes; (void)n_in; (void)out_size; (void)ws_size;
  const float* z_init = (const float*)d_in[0];
  // d_in[1] = num_steps (fixed 64 in this workload)
  const float* w_ih   = (const float*)d_in[2];
  const float* w_hh   = (const float*)d_in[3];
  const float* b_ih   = (const float*)d_in[4];
  const float* b_hh   = (const float*)d_in[5];
  float* out = (float*)d_out;
  bf16* wpack = (bf16*)d_ws;  // needs 2*96*16*32*16*2 = 3 MB of scratch

  const size_t total = (size_t)2 * NTILES_W * KTILES * 32 * 16;
  const int pack_blocks = (int)((total + NTHREADS - 1) / NTHREADS);
  gru_pack_weights<<<pack_blocks, NTHREADS, 0, stream>>>(w_ih, w_hh, wpack);

  gru_recurrence_kernel<<<BATCH / ROWS, NTHREADS, 0, stream>>>(
      z_init, b_ih, b_hh, wpack, out);
}
